// WTA_33990371181558
// MI455X (gfx1250) — matrix operand order repaired and checked by reading, not compiled, not run on hardware
//
#include <hip/hip_runtime.h>
#include <hip/hip_fp16.h>

typedef _Float16 v16h __attribute__((ext_vector_type(16)));
typedef _Float16 v8h  __attribute__((ext_vector_type(8)));
typedef float    v8f  __attribute__((ext_vector_type(8)));

namespace {
constexpr int  kB      = 2;
constexpr int  kDim    = 192;
constexpr int  kH      = 256;
constexpr int  kW      = 256;
constexpr int  kHeads  = 4;
constexpr int  kCl     = 48;               // channels per head
constexpr long kPix    = (long)kH * kW;    // 65536
constexpr int  kFeat   = kCl * kW;         // 12288 (token feature dim)
constexpr long kQkvElems  = (long)kB * 3 * kDim * kPix;          // f32
constexpr long kHeadElems = (long)kB * kHeads * kH * kFeat;      // f16
}

__device__ __forceinline__ v8f wmma_f16(v16h a, v16h b, v8f c) {
  // D(f32 16x16) = A(16x32 f16) * B(32x16 f16) + C
  return __builtin_amdgcn_wmma_f32_16x16x32_f16(false, a, false, b,
                                                (short)0, c, false, false);
}
__device__ __forceinline__ v16h cat8(v8h lo, v8h hi) {
  return __builtin_shufflevector(lo, hi, 0,1,2,3,4,5,6,7,8,9,10,11,12,13,14,15);
}

// --- CDNA5 async global->LDS path (ASYNCcnt), inline asm (no probed builtin).
// LDS byte address = low 32 bits of the generic __shared__ pointer
// (flat->LDS mapping truncates to addr[31:0]).
__device__ __forceinline__ unsigned lds_off(const void* p) {
  return (unsigned)(unsigned long long)p;
}
__device__ __forceinline__ void async_g2l_b128(unsigned lds, const void* g) {
  asm volatile("global_load_async_to_lds_b128 %0, %1, off"
               :: "v"(lds), "v"(g) : "memory");
}
__device__ __forceinline__ void wait_async(int n) {
  if (n) asm volatile("s_wait_asynccnt 0x1" ::: "memory");
  else   asm volatile("s_wait_asynccnt 0x0" ::: "memory");
}

// ---------------------------------------------------------------------------
// Kernel 1: qkv 1x1 conv as GEMM. out[b][oc][p] = sum_ic W[oc][ic]*x[b][ic][p]
// Per wave: one 16(oc) x 16(pixel) f32 tile, K=192 in 6 chunks of 32.
// B tile (X is pixel-contiguous per ic) transposed through per-wave LDS.
// ---------------------------------------------------------------------------
__global__ void __launch_bounds__(256)
k_qkv(const float* __restrict__ x, const float* __restrict__ wq,
      float* __restrict__ qkv) {
  __shared__ __align__(32) _Float16 bst[8 * 16 * 32];
  const int tid = threadIdx.x, lane = tid & 31, wv = tid >> 5;
  const int row = lane & 15, hi = lane >> 4;        // hi selects lane half
  long g = (long)blockIdx.x * 8 + wv;               // wave-tile id
  const int n = (int)(g % (kPix / 16)); g /= (kPix / 16);
  const int m = (int)(g % 36);
  const int b = (int)(g / 36);
  const int  oc0 = m * 16;
  const long p0  = (long)n * 16;
  _Float16* my = &bst[wv * 512];
  const float* xb = x + (long)b * kDim * kPix;

  v8f acc = {};
  for (int k0 = 0; k0 < kDim; k0 += 32) {
    // A fragment from weights (ic-contiguous rows)
    const float* wr = wq + (long)(oc0 + row) * kDim + k0 + hi * 8;
    v16h a;
#pragma unroll
    for (int i = 0; i < 8; ++i) a[i] = (_Float16)wr[i];
#pragma unroll
    for (int i = 0; i < 8; ++i) a[8 + i] = (_Float16)wr[16 + i];
    // B stage: lane owns ic = k0+lane, loads 16 contiguous pixels, transposes
    const float* xr = xb + (long)(k0 + lane) * kPix + p0;
#pragma unroll
    for (int i = 0; i < 16; ++i) my[i * 32 + lane] = (_Float16)xr[i];
    asm volatile("" ::: "memory");                  // keep DS store->load order
    v16h bf = cat8(*(const v8h*)&my[row * 32 + hi * 8], *(const v8h*)&my[row * 32 + hi * 8 + 16]);
    acc = wmma_f16(a, bf, acc);
  }
  float* ob = qkv + ((long)b * (3 * kDim) + oc0) * kPix + p0;
#pragma unroll
  for (int r = 0; r < 8; ++r) ob[(long)(r + hi * 8) * kPix + row] = acc[r];
}

// ---------------------------------------------------------------------------
// Kernel 2: depthwise 3x3 conv + per-token L2 norm -> heads layout f16.
// One block per (b, head, row h). Thread = w column. 48KB LDS row buffer.
// ---------------------------------------------------------------------------
__global__ void __launch_bounds__(256)
k_dwnorm(const float* __restrict__ qkv, const float* __restrict__ wdw,
         _Float16* __restrict__ out, int choff) {
  __shared__ float buf[kCl * kW];
  __shared__ float red[256];
  const int t    = threadIdx.x;                     // w
  const int h    = blockIdx.x % kH;
  const int head = (blockIdx.x / kH) % kHeads;
  const int b    = blockIdx.x / (kH * kHeads);

  float ss = 0.f;
  for (int cl = 0; cl < kCl; ++cl) {
    const int c = head * kCl + cl;
    const float* in = qkv + ((long)b * (3 * kDim) + choff + c) * kPix;
    const float* wk = wdw + c * 9;
    float s = 0.f;
#pragma unroll
    for (int dh = -1; dh <= 1; ++dh) {
      const int h2 = h + dh;
      if ((unsigned)h2 >= (unsigned)kH) continue;
#pragma unroll
      for (int dw = -1; dw <= 1; ++dw) {
        const int w2 = t + dw;
        if ((unsigned)w2 >= (unsigned)kW) continue;
        s += wk[(dh + 1) * 3 + (dw + 1)] * in[(long)h2 * kW + w2];
      }
    }
    buf[cl * kW + t] = s;
    ss += s * s;
  }
  red[t] = ss;
  __syncthreads();
  for (int o = 128; o > 0; o >>= 1) {
    if (t < o) red[t] += red[t + o];
    __syncthreads();
  }
  const float inv = 1.f / fmaxf(sqrtf(red[0]), 1e-12f);
  _Float16* orow = out + (((long)b * kHeads + head) * kH + h) * kFeat;
  for (int cl = 0; cl < kCl; ++cl)
    orow[cl * kW + t] = (_Float16)(buf[cl * kW + t] * inv);
}

// ---------------------------------------------------------------------------
// Kernel 3: MDC value branch. out = sigmoid(v) * grouped_conv_{3,5,7,9}(v),
// written straight into heads layout (head == oc/48 by construction).
// ---------------------------------------------------------------------------
__global__ void __launch_bounds__(256)
k_mdc(const float* __restrict__ qkv,
      const float* __restrict__ wv1, const float* __restrict__ wv2,
      const float* __restrict__ wv3, const float* __restrict__ wv4,
      _Float16* __restrict__ v1out) {
  const long idx = (long)blockIdx.x * 256 + threadIdx.x;
  const int w  = (int)(idx & 255);
  const int h  = (int)((idx >> 8) & 255);
  const int oc = (int)((idx >> 16) % kDim);
  const int b  = (int)(idx / ((long)kDim * kPix));
  const int ci = oc / kCl;                 // which conv (also the head)
  const int gp = oc % kCl;                 // group (also c_local)
  const int ks = 3 + 2 * ci, pad = 1 + ci;
  const float* wv = (ci == 0 ? wv1 : ci == 1 ? wv2 : ci == 2 ? wv3 : wv4)
                    + (long)gp * 4 * ks * ks;
  const float* vin = qkv + ((long)b * (3 * kDim) + 2 * kDim) * kPix;

  float s = 0.f;
  for (int ic = 0; ic < 4; ++ic) {
    const float* ip = vin + (long)(4 * gp + ic) * kPix;
    const float* wk = wv + ic * ks * ks;
    for (int kh = 0; kh < ks; ++kh) {
      const int h2 = h + kh - pad;
      if ((unsigned)h2 >= (unsigned)kH) continue;
      for (int kw = 0; kw < ks; ++kw) {
        const int w2 = w + kw - pad;
        if ((unsigned)w2 >= (unsigned)kW) continue;
        s += wk[kh * ks + kw] * ip[(long)h2 * kW + w2];
      }
    }
  }
  const float vo  = vin[(long)oc * kPix + (long)h * kW + w];
  const float sig = 1.f / (1.f + __expf(-vo));
  v1out[(((long)b * kHeads + ci) * kH + h) * kFeat + gp * kW + w] =
      (_Float16)(sig * s);
}

// ---------------------------------------------------------------------------
// Kernel 4: attention. Block = (b, head, 16-row tile), 8 waves.
// Stage 1: Q tile async-staged global->LDS (double buffered, ASYNCcnt);
//          wave j computes S[16 x 32] cols [32j,32j+32), K=12288 (WMMA).
// Stage 2: row-wise softmax (scaled by temperature) in LDS -> f16 P.
// Stage 3: wave j computes O[16 x 1536] = P @ V over its feature range,
//          V tiles transposed through per-wave LDS (WMMA).
// ---------------------------------------------------------------------------
__global__ void __launch_bounds__(256)
k_attn(const _Float16* __restrict__ q1, const _Float16* __restrict__ k1,
       const _Float16* __restrict__ v1, const float* __restrict__ temp,
       _Float16* __restrict__ o) {
  __shared__ __align__(32) float     S[16 * 256];
  __shared__ __align__(32) _Float16  Pb[16 * 256];
  __shared__ __align__(32) _Float16  bst[8 * 16 * 32];
  __shared__ __align__(32) _Float16  Qs[2][16 * 128];   // double-buffered Q
  const int tid = threadIdx.x, lane = tid & 31, wv = tid >> 5;
  const int row = lane & 15, hi = lane >> 4;
  const int rt   = blockIdx.x % (kH / 16);
  const int head = (blockIdx.x / (kH / 16)) % kHeads;
  const int b    = blockIdx.x / ((kH / 16) * kHeads);
  const long bh  = (long)b * kHeads + head;
  const int  r0  = rt * 16;
  const float tmp = temp[head];

  const _Float16* Q = q1 + (bh * kH + r0) * kFeat;
  const _Float16* K = k1 + bh * kH * kFeat;
  const _Float16* V = v1 + bh * kH * kFeat;

  // ---- Stage 1: S = (Q . K^T) * temperature --------------------------------
  const int col0 = wv * 32;
  const _Float16* Kr0 = K + (long)(col0 + row) * kFeat + hi * 16;
  const _Float16* Kr1 = K + (long)(col0 + 16 + row) * kFeat + hi * 16;
  // each thread async-copies one 16B slice of the 16x128 Q chunk
  const int sl_row = tid >> 4;            // 0..15
  const int sl_col = (tid & 15) * 8;      // half offset 0..120
  const _Float16* Qsrc = Q + (long)sl_row * kFeat + sl_col;
  const unsigned qdst0 = lds_off(&Qs[0][sl_row * 128 + sl_col]);
  const unsigned qdst1 = lds_off(&Qs[1][sl_row * 128 + sl_col]);

  constexpr int KC = 128, NCH = kFeat / KC;   // 96 staged chunks
  async_g2l_b128(qdst0, Qsrc);
  v8f c0 = {}, c1 = {};
  for (int ch = 0; ch < NCH; ++ch) {
    if (ch + 1 < NCH)
      async_g2l_b128(((ch + 1) & 1) ? qdst1 : qdst0, Qsrc + (long)(ch + 1) * KC);
    wait_async(ch + 1 < NCH ? 1 : 0);     // in-order: chunk `ch` has landed
    __syncthreads();                      // all waves' slices visible
    const _Float16* qs = &Qs[ch & 1][0];
#pragma unroll
    for (int s = 0; s < 4; ++s) {
      const int kc = ch * KC + s * 32;
      __builtin_prefetch(Kr0 + kc + 256, 0, 0);
      __builtin_prefetch(Kr1 + kc + 256, 0, 0);
      const _Float16* pr = qs + row * 128 + s * 32 + hi * 16;
      v16h a  = *(const v16h*)pr;
      v16h b0 = *(const v16h*)(Kr0 + kc);
      v16h b1 = *(const v16h*)(Kr1 + kc);
      c0 = wmma_f16(a, b0, c0);
      c1 = wmma_f16(a, b1, c1);
    }
    __syncthreads();                      // buffer free before next overwrite
  }
#pragma unroll
  for (int r = 0; r < 8; ++r) {
    S[(r + hi * 8) * 256 + col0 + row]      = c0[r] * tmp;
    S[(r + hi * 8) * 256 + col0 + 16 + row] = c1[r] * tmp;
  }
  __syncthreads();

  // ---- Stage 2: softmax per token row -------------------------------------
  if (tid < 16) {
    const float* sr = &S[tid * 256];
    float m = -3.4e38f;
    for (int j = 0; j < 256; ++j) m = fmaxf(m, sr[j]);
    float sum = 0.f;
    for (int j = 0; j < 256; ++j) sum += __expf(sr[j] - m);
    const float inv = 1.f / sum;
    for (int j = 0; j < 256; ++j)
      Pb[tid * 256 + j] = (_Float16)(__expf(sr[j] - m) * inv);
  }
  __syncthreads();

  // ---- Stage 3: O = P @ V --------------------------------------------------
  _Float16* my = &bst[wv * 512];
  for (int nt = 0; nt < (kFeat / 8) / 16; ++nt) {     // 96 tiles per wave
    const int n0 = wv * (kFeat / 8) + nt * 16;
    v8f acc = {};
    for (int kt = 0; kt < 8; ++kt) {                  // 8 x 32 tokens
      const _Float16* pr = &Pb[row * 256 + kt * 32 + hi * 8];
      v16h a = cat8(*(const v8h*)pr, *(const v8h*)(pr + 16));
      // B stage: lane owns token kt*32+lane, transpose 16 features into LDS
      v16h vr = *(const v16h*)(V + (long)(kt * 32 + lane) * kFeat + n0);
#pragma unroll
      for (int i = 0; i < 16; ++i) my[i * 32 + lane] = vr[i];
      asm volatile("" ::: "memory");
      v16h bf = cat8(*(const v8h*)&my[row * 32 + hi * 8], *(const v8h*)&my[row * 32 + hi * 8 + 16]);
      acc = wmma_f16(a, bf, acc);
    }
    _Float16* orow = o + (bh * kH + r0) * kFeat + n0;
#pragma unroll
    for (int r = 0; r < 8; ++r)
      orow[(long)(r + hi * 8) * kFeat + row] = (_Float16)acc[r];
  }
}

// ---------------------------------------------------------------------------
// Kernel 5: proj 1x1 conv as GEMM, reading O from heads layout (f16).
// out[b][oc][p] = sum_ic Wp[oc][ic] * O[b, ic/48, h, (ic%48)*W + w]
// ---------------------------------------------------------------------------
__global__ void __launch_bounds__(256)
k_proj(const _Float16* __restrict__ o, const float* __restrict__ wp,
       float* __restrict__ out) {
  __shared__ __align__(32) _Float16 bst[8 * 16 * 32];
  const int tid = threadIdx.x, lane = tid & 31, wv = tid >> 5;
  const int row = lane & 15, hi = lane >> 4;
  long g = (long)blockIdx.x * 8 + wv;
  const int n = (int)(g % (kPix / 16)); g /= (kPix / 16);
  const int m = (int)(g % 12);
  const int b = (int)(g / 12);
  const int  oc0 = m * 16;
  const long p0  = (long)n * 16;
  const int  h   = (int)(p0 >> 8), w0 = (int)(p0 & 255);
  _Float16* my = &bst[wv * 512];

  v8f acc = {};
  for (int k0 = 0; k0 < kDim; k0 += 32) {
    const float* wr = wp + (long)(oc0 + row) * kDim + k0 + hi * 8;
    v16h a;
#pragma unroll
    for (int i = 0; i < 8; ++i) a[i] = (_Float16)wr[i];
#pragma unroll
    for (int i = 0; i < 8; ++i) a[8 + i] = (_Float16)wr[16 + i];
    const int ic = k0 + lane, hd = ic / kCl, cl = ic % kCl;
    const _Float16* orow =
        o + (((long)b * kHeads + hd) * kH + h) * kFeat + cl * kW + w0;
    v16h vr = *(const v16h*)orow;
#pragma unroll
    for (int i = 0; i < 16; ++i) my[i * 32 + lane] = vr[i];
    asm volatile("" ::: "memory");
    v16h bf = cat8(*(const v8h*)&my[row * 32 + hi * 8], *(const v8h*)&my[row * 32 + hi * 8 + 16]);
    acc = wmma_f16(a, bf, acc);
  }
  float* ob = out + ((long)b * kDim + oc0) * kPix + p0;
#pragma unroll
  for (int r = 0; r < 8; ++r) ob[(long)(r + hi * 8) * kPix + row] = acc[r];
}

// ---------------------------------------------------------------------------
extern "C" void kernel_launch(void* const* d_in, const int* in_sizes, int n_in,
                              void* d_out, int out_size, void* d_ws,
                              size_t ws_size, hipStream_t stream) {
  (void)in_sizes; (void)n_in; (void)out_size; (void)ws_size;
  const float* x      = (const float*)d_in[0];
  const float* w_qkv  = (const float*)d_in[1];
  const float* w_q    = (const float*)d_in[2];
  const float* w_k    = (const float*)d_in[3];
  const float* w_v1   = (const float*)d_in[4];
  const float* w_v2   = (const float*)d_in[5];
  const float* w_v3   = (const float*)d_in[6];
  const float* w_v4   = (const float*)d_in[7];
  const float* w_proj = (const float*)d_in[8];
  const float* temp   = (const float*)d_in[9];

  char* ws = (char*)d_ws;
  float*    qkv = (float*)ws;     ws += kQkvElems * sizeof(float);     // 302 MB
  _Float16* q1  = (_Float16*)ws;  ws += kHeadElems * sizeof(_Float16); //  50 MB
  _Float16* k1  = (_Float16*)ws;  ws += kHeadElems * sizeof(_Float16);
  _Float16* v1  = (_Float16*)ws;  ws += kHeadElems * sizeof(_Float16);
  _Float16* ob  = (_Float16*)ws;                                       //  50 MB

  // qkv GEMM: 2 * 36 * 4096 wave tiles / 8 waves per block
  k_qkv<<<(kB * 36 * (int)(kPix / 16)) / 8, 256, 0, stream>>>(x, w_qkv, qkv);
  // depthwise + L2 norm for q and k: block per (b, head, row)
  k_dwnorm<<<kB * kHeads * kH, 256, 0, stream>>>(qkv, w_q, q1, 0);
  k_dwnorm<<<kB * kHeads * kH, 256, 0, stream>>>(qkv, w_k, k1, kDim);
  // MDC value branch: thread per output element
  k_mdc<<<(int)(((long)kB * kDim * kPix) / 256), 256, 0, stream>>>(
      qkv, w_v1, w_v2, w_v3, w_v4, v1);
  // attention: block per (b, head, 16-row tile)
  k_attn<<<kB * kHeads * (kH / 16), 256, 0, stream>>>(q1, k1, v1, temp, ob);
  // proj GEMM: 2 * 12 * 4096 wave tiles / 8 waves per block
  k_proj<<<(kB * 12 * (int)(kPix / 16)) / 8, 256, 0, stream>>>(ob, w_proj,
                                                               (float*)d_out);
}
